// CrossModalGraph_18270790877215
// MI455X (gfx1250) — matrix operand order repaired and checked by reading, not compiled, not run on hardware
//
#include <hip/hip_runtime.h>
#include <cstddef>
#include <cstdint>

typedef __attribute__((ext_vector_type(16))) _Float16 v16h;
typedef __attribute__((ext_vector_type(8)))  _Float16 v8h;
typedef __attribute__((ext_vector_type(8)))  float    v8f;
typedef __attribute__((ext_vector_type(4)))  int      v4i;

#define LMOD  3000
#define NNODE 9000
#define NEDGE 200000
#define DDIM  512
#define SDIM  128

// Async global->LDS path (gfx1250). Fallback: plain LDS stores.
#if defined(__HIP_DEVICE_COMPILE__) && __has_builtin(__builtin_amdgcn_global_load_async_to_lds_b128)
#define CMG_ASYNC_LDS 1
#else
#define CMG_ASYNC_LDS 0
#endif

#if CMG_ASYNC_LDS
typedef __attribute__((address_space(1))) v4i as1_v4i;
typedef __attribute__((address_space(3))) v4i as3_v4i;
#endif

__device__ __forceinline__ float lrelu_f(float x) { return x > 0.f ? x : 0.01f * x; }

__device__ __forceinline__ void atomicAddF32(float* p, float v) {
  __hip_atomic_fetch_add(p, v, __ATOMIC_RELAXED, __HIP_MEMORY_SCOPE_AGENT);
}

// ---------------------------------------------------------------------------
// C[MxN] = act( A[MxK,lda](f32) @ W[NxK,ldw](f16)^T + bias ), fp32 accumulate.
// grid = (ceil(M/64), N/64), block = 128 threads (4 waves).
// Wave: one 16-row tile x 64 cols (4x v8f accumulators).
// B tile (64 cols x 32 K) is fragment-contiguous in global f16; staged to LDS
// via async global->LDS copies (ASYNCcnt) when available.
// ---------------------------------------------------------------------------
__global__ __launch_bounds__(128)
void cmg_gemm_wmma(const float* __restrict__ A, const _Float16* __restrict__ W,
                   const float* __restrict__ bias, float* __restrict__ C,
                   int M_, int N_, int K_, int lda, int ldw, int act)
{
  __shared__ v8h ldsB[4][32][2];   // [n-subtile][lane][lo/hi 8 halfs] = 4KB

  const int tid   = threadIdx.x;
  const int lane  = tid & 31;
  const int wave  = tid >> 5;
  const int mtile = blockIdx.x * 4 + wave;
  const int n0    = blockIdx.y * 64;

  // A-fragment row for this lane (clamped for safe addressing on tail tile)
  int ar = mtile * 16 + (lane & 15);
  if (ar >= M_) ar = M_ - 1;
  const size_t abase = (size_t)ar * lda + ((lane >> 4) << 3);   // + k0 per step

  // B staging: wave w fills subtile w; lane covers col n0+16w+(lane&15),
  // two 8-wide K chunks at 8*(lane>=16) and 8*(lane>=16)+16  (fragment order)
  const int fn = n0 + 16 * wave + (lane & 15);
  const _Float16* wsrc = W + (size_t)fn * ldw + ((lane >> 4) << 3);

  v8f acc[4];
  #pragma unroll
  for (int j = 0; j < 4; ++j) {
    const float bv = bias ? bias[n0 + 16 * j + (lane & 15)] : 0.f;
    #pragma unroll
    for (int v = 0; v < 8; ++v) acc[j][v] = bv;
  }

  for (int k0 = 0; k0 < K_; k0 += 32) {
    // ---- stage B fragment tile into LDS ----
#if CMG_ASYNC_LDS
    {
      __builtin_amdgcn_global_load_async_to_lds_b128(
          (as1_v4i*)(void*)(wsrc + k0),
          (as3_v4i*)(void*)&ldsB[wave][lane][0], 0, 0);
      __builtin_amdgcn_global_load_async_to_lds_b128(
          (as1_v4i*)(void*)(wsrc + k0 + 16),
          (as3_v4i*)(void*)&ldsB[wave][lane][1], 0, 0);
#if __has_builtin(__builtin_amdgcn_s_wait_asynccnt)
      __builtin_amdgcn_s_wait_asynccnt(0);
#else
      asm volatile("s_wait_asynccnt 0" ::: "memory");
#endif
    }
#else
    {
      const v8h* wp = reinterpret_cast<const v8h*>(wsrc + k0);
      ldsB[wave][lane][0] = wp[0];
      ldsB[wave][lane][1] = wp[2];
    }
#endif
    __syncthreads();

    // ---- A fragment: two 8-wide K chunks (k0+sel..+7, k0+sel+16..+23) ----
    v16h a;
    {
      const float4* ap = reinterpret_cast<const float4*>(A + abase + k0);
      const float4 x0 = ap[0], x1 = ap[1], x2 = ap[4], x3 = ap[5];
      a[0]=(_Float16)x0.x;  a[1]=(_Float16)x0.y;  a[2]=(_Float16)x0.z;  a[3]=(_Float16)x0.w;
      a[4]=(_Float16)x1.x;  a[5]=(_Float16)x1.y;  a[6]=(_Float16)x1.z;  a[7]=(_Float16)x1.w;
      a[8]=(_Float16)x2.x;  a[9]=(_Float16)x2.y;  a[10]=(_Float16)x2.z; a[11]=(_Float16)x2.w;
      a[12]=(_Float16)x3.x; a[13]=(_Float16)x3.y; a[14]=(_Float16)x3.z; a[15]=(_Float16)x3.w;
    }
    if (k0 + 32 < K_) __builtin_prefetch(A + abase + k0 + 32, 0, 1);

    #pragma unroll
    for (int j = 0; j < 4; ++j) {
      const v8h blo = ldsB[j][lane][0];
      const v8h bhi = ldsB[j][lane][1];
      const v16h b = __builtin_shufflevector(blo, bhi,
                       0,1,2,3,4,5,6,7,8,9,10,11,12,13,14,15);
      acc[j] = __builtin_amdgcn_wmma_f32_16x16x32_f16(
                 false, a, false, b, (short)0, acc[j], false, false);
    }
    __syncthreads();
  }

  // ---- store per 32-bit C/D layout: row = m0 + v + 8*(lane>=16) ----
  const int rbase = mtile * 16 + ((lane >> 4) << 3);
  #pragma unroll
  for (int j = 0; j < 4; ++j) {
    const int col = n0 + 16 * j + (lane & 15);
    #pragma unroll
    for (int v = 0; v < 8; ++v) {
      const int r = rbase + v;
      if (r < M_) {
        float val = acc[j][v];
        if (act) val = lrelu_f(val);
        C[(size_t)r * N_ + col] = val;
      }
    }
  }
}

// ---------------------------------------------------------------------------
__global__ void cmg_f32_to_f16(const float* __restrict__ s, _Float16* __restrict__ d, int n)
{
  int i = blockIdx.x * blockDim.x + threadIdx.x;
  const int stride = gridDim.x * blockDim.x;
  for (; i < n; i += stride) d[i] = (_Float16)s[i];
}

__global__ void cmg_assemble_feature(const float4* __restrict__ f0,
                                     const float4* __restrict__ f1,
                                     const float4* __restrict__ f2,
                                     float4* __restrict__ x)
{
  const int per = LMOD * DDIM / 4;
  const int i = blockIdx.x * blockDim.x + threadIdx.x;
  if (i >= 3 * per) return;
  float4 v;
  if (i < per)            v = f0[i];
  else if (i < 2 * per)   v = f1[i - per];
  else                    v = f2[i - 2 * per];
  x[i] = v;
}

__global__ void cmg_fill_zero(float* __restrict__ p, int n)
{
  int i = blockIdx.x * blockDim.x + threadIdx.x;
  const int stride = gridDim.x * blockDim.x;
  for (; i < n; i += stride) p[i] = 0.f;
}

// rows of X (nrows x 128): x / max(||x||, 1e-12), one wave per row
__global__ __launch_bounds__(256)
void cmg_l2norm_rows(float* __restrict__ X, int nrows)
{
  const int lane = threadIdx.x & 31;
  const int row  = blockIdx.x * 8 + (threadIdx.x >> 5);
  if (row >= nrows) return;
  float4* rp = reinterpret_cast<float4*>(X + (size_t)row * SDIM);
  float4 v = rp[lane];
  float ss = v.x*v.x + v.y*v.y + v.z*v.z + v.w*v.w;
  #pragma unroll
  for (int o = 16; o > 0; o >>= 1) ss += __shfl_xor(ss, o, 32);
  const float scale = 1.f / fmaxf(sqrtf(ss), 1e-12f);
  v.x *= scale; v.y *= scale; v.z *= scale; v.w *= scale;
  rp[lane] = v;
}

// one wave per edge: w_e = sigmoid(w2 . lrelu(A[src]+B[tgt]) + b2); deg[src] += w_e
__global__ __launch_bounds__(256)
void cmg_edge_weight(const long long* __restrict__ ei,
                     const float* __restrict__ Ab, const float* __restrict__ Bb,
                     const float* __restrict__ w2, const float* __restrict__ b2,
                     float* __restrict__ ew, float* __restrict__ deg)
{
  const int lane = threadIdx.x & 31;
  const int e = blockIdx.x * 8 + (threadIdx.x >> 5);
  if (e >= NEDGE) return;
  const int src = (int)ei[e];
  const int tgt = (int)ei[NEDGE + e];
  const float4 av = reinterpret_cast<const float4*>(Ab + (size_t)src * SDIM)[lane];
  const float4 bv = reinterpret_cast<const float4*>(Bb + (size_t)tgt * SDIM)[lane];
  const float4 wv = reinterpret_cast<const float4*>(w2)[lane];
  float dot = lrelu_f(av.x + bv.x) * wv.x
            + lrelu_f(av.y + bv.y) * wv.y
            + lrelu_f(av.z + bv.z) * wv.z
            + lrelu_f(av.w + bv.w) * wv.w;
  #pragma unroll
  for (int o = 16; o > 0; o >>= 1) dot += __shfl_xor(dot, o, 32);
  if (lane == 0) {
    const float z = dot + b2[0];
    const float s = 1.f / (1.f + __expf(-z));
    ew[e] = s;
    atomicAddF32(&deg[src], s);
  }
}

__global__ void cmg_dinv(const float* __restrict__ deg, float* __restrict__ dinv, int n)
{
  const int i = blockIdx.x * blockDim.x + threadIdx.x;
  if (i < n) {
    const float s = deg[i];
    dinv[i] = s > 0.f ? rsqrtf(s) : 0.f;
  }
}

__global__ void cmg_coef(const long long* __restrict__ ei, const float* __restrict__ ew,
                         const float* __restrict__ dinv, float* __restrict__ coef)
{
  const int e = blockIdx.x * blockDim.x + threadIdx.x;
  if (e < NEDGE) coef[e] = dinv[(int)ei[e]] * ew[e] * dinv[(int)ei[NEDGE + e]];
}

// one wave per edge: nb[src][:] += coef * proj[tgt][:]  (512 channels)
__global__ __launch_bounds__(256)
void cmg_spmm_scatter(const long long* __restrict__ ei, const float* __restrict__ coef,
                      const float* __restrict__ proj, float* __restrict__ nb)
{
  const int lane = threadIdx.x & 31;
  const int e = blockIdx.x * 8 + (threadIdx.x >> 5);
  if (e >= NEDGE) return;
  const int src = (int)ei[e];
  const int tgt = (int)ei[NEDGE + e];
  const float c = coef[e];
  const float* pt = proj + (size_t)tgt * DDIM;
  float* ns = nb + (size_t)src * DDIM;
  #pragma unroll
  for (int i = 0; i < 16; ++i) {
    const int idx = lane + 32 * i;
    atomicAddF32(&ns[idx], c * pt[idx]);
  }
}

// cat = [x + nb, x * nb]  (NNODE x 2D)
__global__ void cmg_cat(const float4* __restrict__ x, const float4* __restrict__ nb,
                        float4* __restrict__ cat)
{
  const int rowq = DDIM / 4;   // 128 float4 per row
  const int i = blockIdx.x * blockDim.x + threadIdx.x;
  if (i >= NNODE * rowq) return;
  const int m = i / rowq, k = i % rowq;
  const float4 xv = x[i], nv = nb[i];
  cat[(size_t)m * 2 * rowq + k] =
      make_float4(xv.x + nv.x, xv.y + nv.y, xv.z + nv.z, xv.w + nv.w);
  cat[(size_t)m * 2 * rowq + rowq + k] =
      make_float4(xv.x * nv.x, xv.y * nv.y, xv.z * nv.z, xv.w * nv.w);
}

// ---------------------------------------------------------------------------
static inline void launch_gemm(const float* A, const _Float16* W, const float* bias,
                               float* C, int M_, int N_, int K_, int lda, int ldw,
                               int act, hipStream_t s)
{
  dim3 grid((M_ + 63) / 64, N_ / 64);
  cmg_gemm_wmma<<<grid, 128, 0, s>>>(A, W, bias, C, M_, N_, K_, lda, ldw, act);
}

static inline void launch_cvt(const float* s, _Float16* d, int n, hipStream_t st)
{
  cmg_f32_to_f16<<<256, 256, 0, st>>>(s, d, n);
}

extern "C" void kernel_launch(void* const* d_in, const int* in_sizes, int n_in,
                              void* d_out, int out_size, void* d_ws, size_t ws_size,
                              hipStream_t stream)
{
  (void)in_sizes; (void)n_in; (void)out_size; (void)ws_size;

  const float*      feat0  = (const float*)d_in[0];
  const float*      feat1  = (const float*)d_in[1];
  const float*      feat2  = (const float*)d_in[2];
  const long long*  ei     = (const long long*)d_in[3];   // int64 (2, E)
  const float*      proj_w = (const float*)d_in[4];
  const float*      mlp_w1 = (const float*)d_in[5];
  const float*      mlp_b1 = (const float*)d_in[6];
  const float*      mlp_w2 = (const float*)d_in[7];
  const float*      mlp_b2 = (const float*)d_in[8];
  const float*      W1f[2] = { (const float*)d_in[9],  (const float*)d_in[11] };
  const float*      W2f[2] = { (const float*)d_in[10], (const float*)d_in[12] };
  float* out = (float*)d_out;
  float* ws  = (float*)d_ws;

  const size_t ND  = (size_t)NNODE * DDIM;        // 4,608,000
  const size_t ND2 = (size_t)NNODE * DDIM * 2;    // 9,216,000
  const size_t NS  = (size_t)NNODE * SDIM;        // 1,152,000

  float* xbuf    = ws;                // feature (N x D)
  float* projb   = xbuf   + ND;       // x @ W1^T
  float* nbb     = projb  + ND;       // gcn @ proj (scatter target)
  float* catb    = nbb    + ND;       // [x+nb, x*nb] (N x 2D)
  float* sharedb = catb   + ND2;      // l2norm(feature @ P^T) (N x S)
  float* Abuf    = sharedb + NS;      // shared @ W1a^T + b1
  float* Bbuf    = Abuf   + NS;       // shared @ W1b^T
  float* ewb     = Bbuf   + NS;       // edge weights (E)
  float* coefb   = ewb    + NEDGE;    // d[src]*w*d[tgt] (E)
  float* degb    = coefb  + NEDGE;    // degrees (N, padded)
  float* dinvb   = degb   + 9216;     // rsqrt degrees

  // f16 weight copies (fragment-ready contiguous K)
  _Float16* wh      = (_Float16*)(dinvb + 9216);
  _Float16* projw_h = wh;                       // 128*512
  _Float16* w1_h    = projw_h + SDIM * DDIM;    // 128*256
  _Float16* W1h0    = w1_h + SDIM * 2 * SDIM;   // 512*512
  _Float16* W1h1    = W1h0 + DDIM * DDIM;       // 512*512
  _Float16* W2h0    = W1h1 + DDIM * DDIM;       // 512*1024
  _Float16* W2h1    = W2h0 + DDIM * 2 * DDIM;   // 512*1024
  const _Float16* W1h[2] = { W1h0, W1h1 };
  const _Float16* W2h[2] = { W2h0, W2h1 };

  // 0) weights -> f16
  launch_cvt(proj_w, projw_h, SDIM * DDIM, stream);
  launch_cvt(mlp_w1, w1_h,    SDIM * 2 * SDIM, stream);
  launch_cvt(W1f[0], W1h0,    DDIM * DDIM, stream);
  launch_cvt(W1f[1], W1h1,    DDIM * DDIM, stream);
  launch_cvt(W2f[0], W2h0,    DDIM * 2 * DDIM, stream);
  launch_cvt(W2f[1], W2h1,    DDIM * 2 * DDIM, stream);

  // 1) assemble contiguous feature matrix
  {
    const int total = NNODE * DDIM / 4;
    cmg_assemble_feature<<<(total + 255) / 256, 256, 0, stream>>>(
        (const float4*)feat0, (const float4*)feat1, (const float4*)feat2,
        (float4*)xbuf);
  }

  // 2) shared = l2norm(feature @ proj_w^T)  (hoisted out of per-edge path)
  launch_gemm(xbuf, projw_h, nullptr, sharedb, NNODE, SDIM, DDIM, DDIM, DDIM, 0, stream);
  cmg_l2norm_rows<<<(NNODE + 7) / 8, 256, 0, stream>>>(sharedb, NNODE);

  // 3) per-node MLP halves: A = shared@W1a^T + b1, B = shared@W1b^T
  launch_gemm(sharedb, w1_h,        mlp_b1,  Abuf, NNODE, SDIM, SDIM, SDIM, 2 * SDIM, 0, stream);
  launch_gemm(sharedb, w1_h + SDIM, nullptr, Bbuf, NNODE, SDIM, SDIM, SDIM, 2 * SDIM, 0, stream);

  // 4) edge weights + degree scatter
  cmg_fill_zero<<<64, 256, 0, stream>>>(degb, 9216);
  cmg_edge_weight<<<(NEDGE + 7) / 8, 256, 0, stream>>>(ei, Abuf, Bbuf, mlp_w2, mlp_b2, ewb, degb);
  cmg_dinv<<<(NNODE + 255) / 256, 256, 0, stream>>>(degb, dinvb, NNODE);
  cmg_coef<<<(NEDGE + 255) / 256, 256, 0, stream>>>(ei, ewb, dinvb, coefb);

  // 5) two graph layers; layer 0 writes x1 into d_out, layer 1 overwrites with final
  const float* xin = xbuf;
  for (int l = 0; l < 2; ++l) {
    launch_gemm(xin, W1h[l], nullptr, projb, NNODE, DDIM, DDIM, DDIM, DDIM, 0, stream);
    cmg_fill_zero<<<1024, 256, 0, stream>>>(nbb, (int)ND);
    cmg_spmm_scatter<<<(NEDGE + 7) / 8, 256, 0, stream>>>(ei, coefb, projb, nbb);
    {
      const int total = NNODE * DDIM / 4;
      cmg_cat<<<(total + 255) / 256, 256, 0, stream>>>(
          (const float4*)xin, (const float4*)nbb, (float4*)catb);
    }
    launch_gemm(catb, W2h[l], nullptr, out, NNODE, DDIM, 2 * DDIM, 2 * DDIM, 2 * DDIM, 1, stream);
    xin = out;
  }
}
